// RSSM_87376814670309
// MI455X (gfx1250) — matrix operand order repaired and checked by reading, not compiled, not run on hardware
//
#include <hip/hip_runtime.h>
#include <math.h>

typedef _Float16 h16;
typedef __attribute__((ext_vector_type(16))) _Float16 v16h;
typedef __attribute__((ext_vector_type(8)))  float    v8f;
typedef __attribute__((ext_vector_type(4)))  unsigned int v4u;

union HFrag { v16h v; v4u u[2]; };

#define TPB 512
#define NWAVES 16

__device__ __forceinline__ float sigmoidf_(float x) { return 1.f / (1.f + expf(-x)); }
__device__ __forceinline__ float softplusf_(float x) { return (x > 20.f) ? x : log1pf(expf(x)); }

// ---------------------------------------------------------------------------
// Async copy global f16 -> LDS, 16 bytes per lane per op (ASYNCcnt tracked).
// elems must be a multiple of 8.
// ---------------------------------------------------------------------------
__device__ __forceinline__ void async_copy_to_lds(h16* dst_lds, const h16* src, int elems)
{
    const int tid = threadIdx.x;
    for (int c = tid; c * 8 < elems; c += TPB) {
        unsigned           laddr = (unsigned)(uintptr_t)(dst_lds + c * 8);
        unsigned long long gaddr = (unsigned long long)(uintptr_t)(src + c * 8);
        asm volatile("global_load_async_to_lds_b128 %0, %1, off"
                     :: "v"(laddr), "v"(gaddr) : "memory");
    }
}

// ---------------------------------------------------------------------------
// One 16xN GEMM phase. A (LDS, f16, row-major [16][a_ld]), WT (LDS or global
// f16, [Npad][w_ld] = transposed weights), C (LDS f32 [16][c_ld]).
// Each wave owns N-tiles wave, wave+16, ... K is consumed in chunks of 32.
// ---------------------------------------------------------------------------
__device__ __forceinline__ void gemm16(const h16* __restrict__ A, int a_ld,
                                       const h16* __restrict__ WT, int w_ld,
                                       int ntiles, int kchunks,
                                       float* __restrict__ C, int c_ld)
{
    const int lane = threadIdx.x & 31;
    const int wave = threadIdx.x >> 5;
    const int m    = lane & 15;
    const int half = lane >> 4;
    const h16* ap = A + m * a_ld + half * 8;            // A frag: K = ko+half*8..+7 and ko+16+half*8..+7
    for (int ti = wave; ti < ntiles; ti += NWAVES) {
        const h16* wp = WT + (size_t)(ti * 16 + m) * w_ld + half * 8;   // B frag: K = ko+half*8..+7 and ko+16+half*8..+7
        v8f acc = {};
        for (int kc = 0; kc < kchunks; ++kc) {
            const int ko = kc * 32;
            HFrag af, bfr;
            af.u[0]  = *(const v4u*)(ap + ko);
            af.u[1]  = *(const v4u*)(ap + ko + 16);
            bfr.u[0] = *(const v4u*)(wp + ko);
            bfr.u[1] = *(const v4u*)(wp + ko + 16);
            acc = __builtin_amdgcn_wmma_f32_16x16x32_f16(
                false, af.v, false, bfr.v, (short)0, acc, false, false);
        }
        float* cp = C + ti * 16 + m;                    // D: row = i + 8*half, col = ti*16 + m
#pragma unroll
        for (int i = 0; i < 8; ++i)
            cp[(i + half * 8) * c_ld] = acc[i];
    }
}

// LayerNorm(200) + SiLU over one row of Cbuf, writes f16 to dst[0..199].
// One wave per row (row = wave id). Uses wave32 shuffles for the reduction.
__device__ __forceinline__ void ln_silu(const float* __restrict__ x,
                                        const float* __restrict__ s,
                                        const float* __restrict__ b,
                                        h16* __restrict__ dst)
{
    const int lane = threadIdx.x & 31;
    float sum = 0.f, sq = 0.f;
    for (int i = lane; i < 200; i += 32) { float v = x[i]; sum += v; sq += v * v; }
#pragma unroll
    for (int o = 16; o; o >>= 1) { sum += __shfl_xor(sum, o); sq += __shfl_xor(sq, o); }
    const float mean = sum * (1.f / 200.f);
    const float rstd = rsqrtf(sq * (1.f / 200.f) - mean * mean + 1e-3f);
    for (int i = lane; i < 200; i += 32) {
        float y = (x[i] - mean) * rstd * s[i] + b[i];
        dst[i] = (h16)(y * (1.f / (1.f + expf(-y))));
    }
}

// ---------------------------------------------------------------------------
// Prep: convert f32 weight [K][N] -> padded transposed f16 [Npad][Kpad]
// ---------------------------------------------------------------------------
__global__ void wconv(const float* __restrict__ src, h16* __restrict__ dst,
                      int K, int N, int Kp, int Np)
{
    int idx = blockIdx.x * blockDim.x + threadIdx.x;
    if (idx >= Kp * Np) return;
    int n = idx / Kp, k = idx % Kp;
    float v = (n < N && k < K) ? src[(size_t)k * N + n] : 0.f;
    dst[idx] = (h16)v;
}

// Prep: initial carry  deter0 = tanh(w_init);  stoch0 = mean-part of
// silu(LN(deter0 @ w_imgout)) @ w_ims + b_ims
__global__ void init_state(const float* __restrict__ w_init,
                           const float* __restrict__ w_imgout,
                           const float* __restrict__ ln_s,
                           const float* __restrict__ ln_b,
                           const float* __restrict__ w_ims,
                           const float* __restrict__ b_ims,
                           float* __restrict__ d0, float* __restrict__ s0)
{
    __shared__ float dd[200];
    __shared__ float t0[200];
    __shared__ float mv[2];
    int tid = threadIdx.x;
    if (tid < 200) dd[tid] = tanhf(w_init[tid]);
    __syncthreads();
    if (tid < 200) {
        float acc = 0.f;
        for (int k = 0; k < 200; ++k) acc += dd[k] * w_imgout[k * 200 + tid];
        t0[tid] = acc;
    }
    __syncthreads();
    if (tid == 0) {
        float s = 0.f, sq = 0.f;
        for (int i = 0; i < 200; ++i) { s += t0[i]; sq += t0[i] * t0[i]; }
        float m = s / 200.f;
        mv[0] = m;
        mv[1] = rsqrtf(sq / 200.f - m * m + 1e-3f);
    }
    __syncthreads();
    if (tid < 200) {
        float y = (t0[tid] - mv[0]) * mv[1] * ln_s[tid] + ln_b[tid];
        t0[tid] = y * (1.f / (1.f + expf(-y)));
        d0[tid] = dd[tid];
    }
    __syncthreads();
    if (tid < 30) {
        float acc = b_ims[tid];
        for (int k = 0; k < 200; ++k) acc += t0[k] * w_ims[k * 60 + tid];
        s0[tid] = acc;
    }
}

// ---------------------------------------------------------------------------
// Main persistent scan kernel: 32 blocks x 16 batch rows, 512 threads (16 waves)
// Small weight matrices are cached in LDS via async-to-LDS copies; the two
// large ones (w_gru, w_obsout) stream from L2 (192MB, weights stay resident).
// ---------------------------------------------------------------------------
__global__ __launch_bounds__(TPB)
void rssm_scan(const float* __restrict__ embed, const float* __restrict__ action,
               const float* __restrict__ is_first,
               const float* __restrict__ nzp, const float* __restrict__ nzq,
               const h16* __restrict__ WTinp, const h16* __restrict__ WTgru,
               const h16* __restrict__ WTimg, const h16* __restrict__ WTobsout,
               const h16* __restrict__ WTims, const h16* __restrict__ WTobs,
               const float* __restrict__ ln_inp_s, const float* __restrict__ ln_inp_b,
               const float* __restrict__ ln_img_s, const float* __restrict__ ln_img_b,
               const float* __restrict__ ln_obs_s, const float* __restrict__ ln_obs_b,
               const float* __restrict__ b_ims, const float* __restrict__ b_obs,
               const float* __restrict__ d0g, const float* __restrict__ s0g,
               float* __restrict__ out)
{
    __shared__ __align__(16) h16   Abuf[16 * 1248];   // generic A operand (max K = 1248)
    __shared__ __align__(16) float Cbuf[16 * 608];    // GEMM f32 output (max N = 608)
    __shared__ __align__(16) float deterF[16 * 200];
    __shared__ __align__(16) float stochF[16 * 30];
    __shared__ __align__(16) h16   deterB[16 * 224];  // padded f16 deter (A for imgout)
    __shared__ __align__(16) h16   hB[16 * 224];      // padded f16 hidden (A for ims/obs)
    // LDS-resident small weights (transposed/padded, f16)
    __shared__ __align__(16) h16   LWinp[208 * 64];
    __shared__ __align__(16) h16   LWimg[208 * 224];
    __shared__ __align__(16) h16   LWims[64 * 224];
    __shared__ __align__(16) h16   LWobs[64 * 224];
    __shared__ float d0[200];
    __shared__ float s0[32];
    __shared__ float fb[16];

    const int tid   = threadIdx.x;
    const int wave  = tid >> 5;
    const int bbase = blockIdx.x * 16;

    // kick off async weight staging first so it overlaps the state init
    async_copy_to_lds(LWinp, WTinp, 208 * 64);
    async_copy_to_lds(LWimg, WTimg, 208 * 224);
    async_copy_to_lds(LWims, WTims, 64 * 224);
    async_copy_to_lds(LWobs, WTobs, 64 * 224);

    for (int i = tid; i < 200; i += TPB) d0[i] = d0g[i];
    for (int i = tid; i < 30;  i += TPB) s0[i] = s0g[i];
    __syncthreads();
    for (int i = tid; i < 16 * 200; i += TPB) deterF[i] = d0[i % 200];
    for (int i = tid; i < 16 * 30;  i += TPB) stochF[i] = s0[i % 30];
    for (int i = tid; i < 16 * 224; i += TPB) {
        int n = i % 224;
        deterB[i] = (n < 200) ? (h16)d0[n] : (h16)0.f;
        hB[i]     = (h16)0.f;
    }
    asm volatile("s_wait_asynccnt 0x0" ::: "memory");
    __syncthreads();

    for (int t = 0; t < 64; ++t) {
        const float* emb_t = embed    + (size_t)t * 512 * 1024 + (size_t)bbase * 1024;
        const float* act_t = action   + (size_t)t * 512 * 6    + bbase * 6;
        const float* isf_t = is_first + (size_t)t * 512        + bbase;
        const float* nzp_t = nzp      + (size_t)t * 512 * 30   + bbase * 30;
        const float* nzq_t = nzq      + (size_t)t * 512 * 30   + bbase * 30;
        float*       out_t = out      + (size_t)t * 512 * 380  + (size_t)bbase * 380;

        // ---- phase 1: is_first reset + build inp A (K padded to 64) ----
        if (tid < 16) fb[tid] = isf_t[tid];
        __syncthreads();
        for (int i = tid; i < 16 * 200; i += TPB) {
            int r = i / 200, n = i % 200;
            float f = fb[r];
            deterF[i] = deterF[i] * (1.f - f) + d0[n] * f;
        }
        for (int i = tid; i < 16 * 64; i += TPB) {
            int r = i >> 6, k = i & 63;
            float f = fb[r], v;
            if (k < 30) {
                float s = stochF[r * 30 + k] * (1.f - f) + s0[k] * f;
                stochF[r * 30 + k] = s;
                v = s;
            } else if (k < 36) {
                float a = act_t[r * 6 + (k - 30)];
                v = a * (1.f / fmaxf(fabsf(a), 1.f)) * (1.f - f);
            } else v = 0.f;
            Abuf[r * 64 + k] = (h16)v;
        }
        __syncthreads();

        // ---- x_pre = [stoch,a] @ w_inp  (K=64, N=208) ---- (LDS weights)
        gemm16(Abuf, 64, LWinp, 64, 13, 2, Cbuf, 608);
        __syncthreads();

        // ---- LN+SiLU -> x ; build GRU A = [x | deter] (K padded 416) ----
        {
            const int r = wave, lane = tid & 31;
            ln_silu(Cbuf + r * 608, ln_inp_s, ln_inp_b, Abuf + r * 416);
            for (int n = lane; n < 200; n += 32)
                Abuf[r * 416 + 200 + n] = (h16)deterF[r * 200 + n];
            if (lane < 16) Abuf[r * 416 + 400 + lane] = (h16)0.f;
        }
        __syncthreads();

        // ---- parts = [x,deter] @ w_gru  (K=416, N=608) ---- (L2 stream)
        gemm16(Abuf, 416, WTgru, 416, 38, 13, Cbuf, 608);
        __syncthreads();

        // ---- GRU gate math -> new deter ----
        for (int i = tid; i < 16 * 200; i += TPB) {
            int r = i / 200, n = i % 200;
            float rr = sigmoidf_(Cbuf[r * 608 + n]);
            float cc = tanhf(rr * Cbuf[r * 608 + 200 + n]);
            float uu = sigmoidf_(Cbuf[r * 608 + 400 + n] - 1.f);
            float d  = uu * cc + (1.f - uu) * deterF[i];
            deterF[i] = d;
            deterB[r * 224 + n] = (h16)d;
        }
        __syncthreads();

        // ---- hp_pre = deter @ w_imgout  (K=224, N=208) ---- (LDS weights)
        gemm16(deterB, 224, LWimg, 224, 13, 7, Cbuf, 608);
        __syncthreads();
        { ln_silu(Cbuf + wave * 608, ln_img_s, ln_img_b, hB + wave * 224); }
        __syncthreads();

        // ---- prior stats = hp @ w_ims  (K=224, N=64) ---- (LDS weights)
        gemm16(hB, 224, LWims, 224, 4, 7, Cbuf, 608);
        __syncthreads();

        // ---- prior outputs + deter output + build obs A = [deter|embed] ----
        for (int i = tid; i < 16 * 30; i += TPB) {
            int r = i / 30, n = i % 30;
            float pm = Cbuf[r * 608 + n] + b_ims[n];
            float ps = softplusf_(Cbuf[r * 608 + 30 + n] + b_ims[30 + n]) + 0.1f;
            float pr = pm + ps * nzp_t[r * 30 + n];
            float* o = out_t + r * 380;
            o[90 + n] = pm; o[120 + n] = ps; o[150 + n] = pr;
        }
        for (int i = tid; i < 16 * 200; i += TPB) {
            int r = i / 200, n = i % 200;
            out_t[r * 380 + 180 + n] = deterF[i];
            Abuf[r * 1248 + n] = deterB[r * 224 + n];
        }
        for (int i = tid; i < 16 * 1024; i += TPB) {
            int r = i >> 10, n = i & 1023;
            Abuf[r * 1248 + 200 + n] = (h16)emb_t[r * 1024 + n];
        }
        for (int i = tid; i < 16 * 24; i += TPB) {
            int r = i / 24, n = i % 24;
            Abuf[r * 1248 + 1224 + n] = (h16)0.f;
        }
        __syncthreads();

        // ---- ho_pre = [deter,embed] @ w_obsout  (K=1248, N=208) ---- (L2 stream)
        gemm16(Abuf, 1248, WTobsout, 1248, 13, 39, Cbuf, 608);
        __syncthreads();
        { ln_silu(Cbuf + wave * 608, ln_obs_s, ln_obs_b, hB + wave * 224); }
        __syncthreads();

        // ---- post stats = ho @ w_obs  (K=224, N=64) ---- (LDS weights)
        gemm16(hB, 224, LWobs, 224, 4, 7, Cbuf, 608);
        __syncthreads();

        for (int i = tid; i < 16 * 30; i += TPB) {
            int r = i / 30, n = i % 30;
            float qm = Cbuf[r * 608 + n] + b_obs[n];
            float qs = softplusf_(Cbuf[r * 608 + 30 + n] + b_obs[30 + n]) + 0.1f;
            float po = qm + qs * nzq_t[r * 30 + n];
            float* o = out_t + r * 380;
            o[n] = qm; o[30 + n] = qs; o[60 + n] = po;
            stochF[i] = po;   // carry
        }
        __syncthreads();
    }
}

// ---------------------------------------------------------------------------
extern "C" void kernel_launch(void* const* d_in, const int* in_sizes, int n_in,
                              void* d_out, int out_size, void* d_ws, size_t ws_size,
                              hipStream_t stream)
{
    const float* embed    = (const float*)d_in[0];
    const float* action   = (const float*)d_in[1];
    const float* is_first = (const float*)d_in[2];
    const float* nzp      = (const float*)d_in[3];
    const float* nzq      = (const float*)d_in[4];
    const float* w_inp    = (const float*)d_in[5];
    const float* ln_inp_s = (const float*)d_in[6];
    const float* ln_inp_b = (const float*)d_in[7];
    const float* w_gru    = (const float*)d_in[8];
    const float* w_imgout = (const float*)d_in[9];
    const float* ln_img_s = (const float*)d_in[10];
    const float* ln_img_b = (const float*)d_in[11];
    const float* w_obsout = (const float*)d_in[12];
    const float* ln_obs_s = (const float*)d_in[13];
    const float* ln_obs_b = (const float*)d_in[14];
    const float* w_ims    = (const float*)d_in[15];
    const float* b_ims    = (const float*)d_in[16];
    const float* w_obs    = (const float*)d_in[17];
    const float* b_obs    = (const float*)d_in[18];
    const float* w_init   = (const float*)d_in[19];

    char* ws = (char*)d_ws;
    size_t off = 0;
    auto take = [&](size_t bytes) -> size_t {
        size_t o = off;
        off += (bytes + 255) & ~(size_t)255;
        return o;
    };
    h16*   WTinp    = (h16*)(ws + take((size_t)208 * 64   * 2));
    h16*   WTgru    = (h16*)(ws + take((size_t)608 * 416  * 2));
    h16*   WTimg    = (h16*)(ws + take((size_t)208 * 224  * 2));
    h16*   WTobsout = (h16*)(ws + take((size_t)208 * 1248 * 2));
    h16*   WTims    = (h16*)(ws + take((size_t)64  * 224  * 2));
    h16*   WTobs    = (h16*)(ws + take((size_t)64  * 224  * 2));
    float* d0       = (float*)(ws + take(200 * 4));
    float* s0       = (float*)(ws + take(32 * 4));

    auto conv = [&](const float* src, h16* dst, int K, int N, int Kp, int Np) {
        int tot = Kp * Np;
        wconv<<<(tot + 255) / 256, 256, 0, stream>>>(src, dst, K, N, Kp, Np);
    };
    conv(w_inp,    WTinp,    36,   200, 64,   208);
    conv(w_gru,    WTgru,    400,  600, 416,  608);
    conv(w_imgout, WTimg,    200,  200, 224,  208);
    conv(w_obsout, WTobsout, 1224, 200, 1248, 208);
    conv(w_ims,    WTims,    200,  60,  224,  64);
    conv(w_obs,    WTobs,    200,  60,  224,  64);

    init_state<<<1, 256, 0, stream>>>(w_init, w_imgout, ln_img_s, ln_img_b,
                                      w_ims, b_ims, d0, s0);

    rssm_scan<<<32, TPB, 0, stream>>>(embed, action, is_first, nzp, nzq,
                                      WTinp, WTgru, WTimg, WTobsout, WTims, WTobs,
                                      ln_inp_s, ln_inp_b, ln_img_s, ln_img_b,
                                      ln_obs_s, ln_obs_b, b_ims, b_obs,
                                      d0, s0, (float*)d_out);
}
